// VectorQuantizer_29609504539291
// MI455X (gfx1250) — matrix operand order repaired and checked by reading, compiled, not run on hardware
//
#include <hip/hip_runtime.h>

#define DIM   256
#define NEMB  16384
#define NPTS  16384
#define ROWS  64            // rows of z per block
#define CPI   32            // codes per iteration (2 col tiles of 16)
#define LDA   (DIM + 8)     // padded LDS row stride (bf16 units) -> 528B, 16B aligned
#define NIT   (NEMB / CPI)

typedef __attribute__((ext_vector_type(16))) __bf16 v16bf;
typedef __attribute__((ext_vector_type(8)))  float  v8f;
typedef int v4i __attribute__((vector_size(16)));
typedef __attribute__((address_space(1))) v4i gv4i;   // global int4
typedef __attribute__((address_space(3))) v4i lv4i;   // LDS int4

union Frag  { uint4 q[2]; v16bf v; };
union CFrag { float f[8];  v8f  v; };

#if defined(__has_builtin)
#if __has_builtin(__builtin_amdgcn_global_load_async_to_lds_b128)
#define HAVE_ASYNC_LDS 1
#endif
#endif
#ifndef HAVE_ASYNC_LDS
#define HAVE_ASYNC_LDS 0
#endif

static __device__ __forceinline__ void async_wait_le4() {
#if defined(__has_builtin) && __has_builtin(__builtin_amdgcn_s_wait_asynccnt)
    __builtin_amdgcn_s_wait_asynccnt(4);
#else
    asm volatile("s_wait_asynccnt 0x4" ::: "memory");
#endif
}
static __device__ __forceinline__ void async_wait_le0() {
#if defined(__has_builtin) && __has_builtin(__builtin_amdgcn_s_wait_asynccnt)
    __builtin_amdgcn_s_wait_asynccnt(0);
#else
    asm volatile("s_wait_asynccnt 0x0" ::: "memory");
#endif
}

static __device__ __forceinline__ void async_cp16(const unsigned short* g, unsigned short* l) {
#if HAVE_ASYNC_LDS
    __builtin_amdgcn_global_load_async_to_lds_b128((gv4i*)g, (lv4i*)l, 0, 0);
#endif
}

static __device__ __forceinline__ unsigned short f2bf(float f) {
    unsigned u = __float_as_uint(f);
    u += 0x7FFFu + ((u >> 16) & 1u);     // round-to-nearest-even
    return (unsigned short)(u >> 16);
}

// ---------------------------------------------------------------------------
// Kernel 1: codebook -> bf16 copy + per-code squared norms. One wave per code.
// ---------------------------------------------------------------------------
__global__ __launch_bounds__(256) void vq_prep(const float* __restrict__ cb,
                                               unsigned short* __restrict__ cbbf,
                                               float* __restrict__ norms) {
    const int code = blockIdx.x * 8 + (threadIdx.x >> 5);
    const int lane = threadIdx.x & 31;
    const float* row = cb + (size_t)code * DIM;
    unsigned short* brow = cbbf + (size_t)code * DIM;
    float s = 0.f;
    #pragma unroll
    for (int i = lane * 8; i < lane * 8 + 8; i += 4) {
        float4 v = *(const float4*)(row + i);
        s += v.x * v.x + v.y * v.y + v.z * v.z + v.w * v.w;
        ushort4 p;
        p.x = f2bf(v.x); p.y = f2bf(v.y); p.z = f2bf(v.z); p.w = f2bf(v.w);
        *(ushort4*)(brow + i) = p;
    }
    #pragma unroll
    for (int off = 16; off >= 1; off >>= 1) s += __shfl_xor(s, off, 32);
    if (lane == 0) norms[code] = s;
}

// ---------------------------------------------------------------------------
// Kernel 2: distance GEMM on WMMA + running argmin + gather.
// 256 blocks x 256 threads (8 waves). Wave w: rowTile = w>>1, colSel = w&1.
// ---------------------------------------------------------------------------
__global__ __launch_bounds__(256) void vq_main(const float* __restrict__ z,
                                               const float* __restrict__ cb,
                                               const unsigned short* __restrict__ cbbf,
                                               const float* __restrict__ norms,
                                               float* __restrict__ zq,
                                               int* __restrict__ idxOut) {
    // phase 1: zT[64][LDA]  (33792B) ; phase 2: two B buffers of 32*LDA (2 x 16896B)
    __shared__ unsigned short smem[ROWS * LDA];
    __shared__ float redD[ROWS][2];
    __shared__ int   redI[ROWS][2];
    __shared__ int   bestRow[ROWS];

    const int tid  = threadIdx.x;
    const int wave = tid >> 5;
    const int lane = tid & 31;
    const int n0   = blockIdx.x * ROWS;
    const int b    = n0 >> 10;          // 1024 spatial points per batch image
    const int hw0  = n0 & 1023;

    // ---- stage transposed z tile (bf16) : zT[row][c] = z[b, c, hw0+row] ----
    {
        const float* zb = z + (size_t)b * DIM * 1024 + hw0;
        #pragma unroll 4
        for (int it = 0; it < 64; ++it) {
            int c = it * 4 + (tid >> 6);     // 4 channels per pass, coalesced in hw
            int j = tid & 63;
            smem[j * LDA + c] = f2bf(zb[(size_t)c * 1024 + j]);
        }
    }
    __syncthreads();

    const int rowTile = wave >> 1;     // 0..3
    const int colSel  = wave & 1;      // 0..1
    const int rlane   = lane & 15;
    const int hi      = lane >> 4;     // lane half selects K sub-range per ISA layout

    // ---- A panel resident in registers: 8 k-step fragments (16x32 bf16 each) ----
    Frag A[8];
    {
        const unsigned short* zr = &smem[(rowTile * 16 + rlane) * LDA];
        #pragma unroll
        for (int k = 0; k < 8; ++k) {
            // dwords 0-3: K = k*32 + hi*8 + 0..7 ; dwords 4-7: K = k*32 + 16 + hi*8 + 0..7
            A[k].q[0] = *(const uint4*)(zr + k * 32 + hi * 8);
            A[k].q[1] = *(const uint4*)(zr + k * 32 + 16 + hi * 8);
        }
    }
    __syncthreads();                   // zT dead; reuse smem as double B buffer

    float bestD[8];
    int   bestI[8];
    #pragma unroll
    for (int i = 0; i < 8; ++i) { bestD[i] = 3.4e38f; bestI[i] = 0; }

    unsigned short* bT0 = smem;
    unsigned short* bT1 = smem + CPI * LDA;
    const int myCodeOff = colSel * 16 + rlane;
    const int crow = tid >> 3, seg = tid & 7;   // staging: 32 codes x 512B, 64B/thread

#if HAVE_ASYNC_LDS
    {   // prologue: async-stage tile 0 into buffer 0 (4 x b128 per thread)
        const unsigned short* src = cbbf + (size_t)crow * DIM + seg * 32;
        unsigned short* dst = bT0 + crow * LDA + seg * 32;
        #pragma unroll
        for (int q = 0; q < 4; ++q) async_cp16(src + q * 8, dst + q * 8);
    }
#endif

    for (int it = 0; it < NIT; ++it) {
        unsigned short* cur = (it & 1) ? bT1 : bT0;
#if HAVE_ASYNC_LDS
        if (it + 1 < NIT) {            // stream next tile while computing this one
            const unsigned short* src = cbbf + (size_t)(it + 1) * CPI * DIM
                                             + (size_t)crow * DIM + seg * 32;
            unsigned short* dst = ((it & 1) ? bT0 : bT1) + crow * LDA + seg * 32;
            #pragma unroll
            for (int q = 0; q < 4; ++q) async_cp16(src + q * 8, dst + q * 8);
            async_wait_le4();          // my tile-it loads landed (in-order completion)
        } else {
            async_wait_le0();
        }
        __syncthreads();               // everyone's tile-it loads landed
#else
        {   // fallback: synchronous staging through VGPRs
            const uint4* src = (const uint4*)(cbbf + (size_t)(it * CPI + crow) * DIM);
            uint4* dst = (uint4*)(cur + crow * LDA);
            #pragma unroll
            for (int q = 0; q < 4; ++q) dst[seg * 4 + q] = src[seg * 4 + q];
        }
        if (it + 1 < NIT)
            __builtin_prefetch(cbbf + (size_t)(it + 1) * CPI * DIM + tid * 32, 0, 1);
        __syncthreads();
#endif

        CFrag acc0, acc1;              // two independent XDL accumulation chains
        #pragma unroll
        for (int i = 0; i < 8; ++i) { acc0.f[i] = 0.f; acc1.f[i] = 0.f; }

        const unsigned short* brow = cur + myCodeOff * LDA;
        #pragma unroll
        for (int k = 0; k < 8; k += 2) {
            Frag B0, B1;               // B 32x16: dwords 0-3 K=hi*8+0..7, dwords 4-7 K=16+hi*8+0..7
            B0.q[0] = *(const uint4*)(brow + k * 32 + hi * 8);
            B0.q[1] = *(const uint4*)(brow + k * 32 + 16 + hi * 8);
            B1.q[0] = *(const uint4*)(brow + (k + 1) * 32 + hi * 8);
            B1.q[1] = *(const uint4*)(brow + (k + 1) * 32 + 16 + hi * 8);
            acc0.v = __builtin_amdgcn_wmma_f32_16x16x32_bf16(
                         false, A[k].v,     false, B0.v, (short)0, acc0.v, false, false);
            acc1.v = __builtin_amdgcn_wmma_f32_16x16x32_bf16(
                         false, A[k + 1].v, false, B1.v, (short)0, acc1.v, false, false);
        }

        const int  code = it * CPI + myCodeOff;
        const float nrm = norms[code];
        #pragma unroll
        for (int i = 0; i < 8; ++i) {
            float d = nrm - 2.0f * (acc0.f[i] + acc1.f[i]);  // ||e||^2 - 2 z.e
            if (d < bestD[i]) { bestD[i] = d; bestI[i] = code; }
        }
        __syncthreads();               // frees cur for overwrite next iteration
    }

    // ---- argmin reduce across the 16 N-lanes (per lane-half) ----
    #pragma unroll
    for (int i = 0; i < 8; ++i) {
        float d = bestD[i]; int ix = bestI[i];
        #pragma unroll
        for (int off = 8; off >= 1; off >>= 1) {
            float od = __shfl_xor(d, off, 32);
            int   oi = __shfl_xor(ix, off, 32);
            if (od < d || (od == d && oi < ix)) { d = od; ix = oi; }
        }
        bestD[i] = d; bestI[i] = ix;
    }
    if (rlane == 0) {
        #pragma unroll
        for (int i = 0; i < 8; ++i) {
            int row = rowTile * 16 + hi * 8 + i;   // C layout: VGPR i -> M = i + 8*hi
            redD[row][colSel] = bestD[i];
            redI[row][colSel] = bestI[i];
        }
    }
    __syncthreads();

    // ---- combine the 2 col-waves, emit idx ----
    if (tid < ROWS) {
        float d0 = redD[tid][0], d1 = redD[tid][1];
        int   i0 = redI[tid][0], i1 = redI[tid][1];
        int ix = (d1 < d0 || (d1 == d0 && i1 < i0)) ? i1 : i0;
        bestRow[tid] = ix;
        idxOut[n0 + tid] = ix;
    }
    __syncthreads();

    // ---- z_q gather from exact fp32 codebook, coalesced float4 ----
    for (int chunk = tid; chunk < ROWS * (DIM / 4); chunk += 256) {
        int row = chunk >> 6;          // 64 float4 per row
        int q   = chunk & 63;
        const float4* src = (const float4*)(cb + (size_t)bestRow[row] * DIM);
        float4* dst = (float4*)(zq + (size_t)(n0 + row) * DIM);
        dst[q] = src[q];
    }
}

extern "C" void kernel_launch(void* const* d_in, const int* in_sizes, int n_in,
                              void* d_out, int out_size, void* d_ws, size_t ws_size,
                              hipStream_t stream) {
    const float* z  = (const float*)d_in[0];   // [16,256,32,32]
    const float* cb = (const float*)d_in[1];   // [16384,256]

    unsigned short* cbbf = (unsigned short*)d_ws;                       // 8 MB
    float* norms = (float*)((char*)d_ws + (size_t)NEMB * DIM * sizeof(unsigned short)); // 64 KB

    float* zq  = (float*)d_out;                                // [16384,256] fp32
    int* idxOut = (int*)((float*)d_out + (size_t)NPTS * DIM);  // [16384] int32

    vq_prep<<<NEMB / 8, 256, 0, stream>>>(cb, cbbf, norms);
    vq_main<<<NPTS / ROWS, 256, 0, stream>>>(z, cb, cbbf, norms, zq, idxOut);
}
